// LRModel_27255862460429
// MI455X (gfx1250) — matrix operand order repaired and checked by reading, not compiled, not run on hardware
//
#include <hip/hip_runtime.h>
#include <hip/hip_bf16.h>
#include <math.h>

// ---------------------------------------------------------------------------
// LR model forward: logits = gate[2048x65536] @ bias[65536] + gbias;
//                   pred = sigmoid(logits); loss = sum(BCE-with-logits)
// Memory-bound GEMV (512MB stream @ 23.3TB/s ~ 22us floor). WMMA f16 is used
// as the accumulation engine: gate is exactly {0,1} (f16-exact), bias ~1e-2
// (f16 rel err ~5e-4), f32 accumulation via V_WMMA_F32_16X16X32_F16.
// ---------------------------------------------------------------------------

typedef __attribute__((ext_vector_type(16))) _Float16 v16h;
typedef __attribute__((ext_vector_type(8)))  float    v8f;
typedef __attribute__((ext_vector_type(2)))  __fp16   fp16x2;

#define BATCH   2048
#define FID     65536
#define ROWS_WG 16
#define SPLITK  16
#define KCHUNK  (FID / SPLITK)     // 4096 columns per workgroup
#define TILE_K  512                // columns staged in LDS per step
#define NTILES  (KCHUNK / TILE_K)  // 8
#define SP      260                // LDS row stride in dwords (256 data + 4 pad)

// Single v_cvt_pk_rtz_f16_f32: packs (lo -> [15:0], hi -> [31:16]) in one op.
__device__ __forceinline__ unsigned pk_f16(float lo, float hi) {
    union { fp16x2 h; unsigned u; } v;
    v.h = __builtin_amdgcn_cvt_pkrtz(lo, hi);
    return v.u;
}

__global__ __launch_bounds__(128) void lr_gemv_wmma(
    const float* __restrict__ gate, const float* __restrict__ bias,
    float* __restrict__ part)
{
    // 16 rows x 512 cols of gate as packed f16 pairs; stride 260 dwords so the
    // per-lane b128 A-fragment reads (start bank = 4*m + const) are conflict-free.
    __shared__ unsigned sgd[ROWS_WG * SP];
    __shared__ unsigned sbd[TILE_K / 2];
    __shared__ float    spart[4][ROWS_WG];

    const int t    = threadIdx.x;     // 0..127 (4 waves of 32)
    const int wave = t >> 5;
    const int lane = t & 31;
    const int h    = lane >> 4;       // lane half
    const int m    = lane & 15;       // matrix row within fragment
    const int rowbase = blockIdx.x * ROWS_WG;
    const int kbase   = blockIdx.y * KCHUNK;

    v8f acc = {};                     // 16x16 f32 accumulator (all N columns equal)

    for (int tile = 0; tile < NTILES; ++tile) {
        const int kt = kbase + tile * TILE_K;

        // ---- stage gate tile: iteration r streams one full 2KB row segment
        //      with 128 lanes * b128 -> perfectly coalesced; convert to f16.
        for (int r = 0; r < ROWS_WG; ++r) {
            const size_t roff = (size_t)(rowbase + r) * FID + kt;
            const float4 g = *(const float4*)(gate + roff + t * 4);
            if (tile + 1 < NTILES)   // speculative prefetch of next tile (global_prefetch_b8)
                __builtin_prefetch(gate + roff + TILE_K + t * 4, 0, 1);
            *(uint2*)&sgd[r * SP + t * 2] = make_uint2(pk_f16(g.x, g.y), pk_f16(g.z, g.w));
        }
        // ---- stage bias tile (512 f32 -> 256 packed dwords)
        {
            const float4 b4 = *(const float4*)(bias + kt + t * 4);
            *(uint2*)&sbd[t * 2] = make_uint2(pk_f16(b4.x, b4.y), pk_f16(b4.z, b4.w));
        }
        __syncthreads();

        // ---- each wave owns a 128-column strip: 4 WMMA steps of K=32.
        const int cw2 = wave * 64;               // strip offset in dwords
        #pragma unroll
        for (int it = 0; it < 4; ++it) {
            const int cb2  = cw2 + it * 16;      // 32-column group base (dwords)
            // A (16x32 f16): lane(m,h) VGPRs 0-3 = K 8h..8h+7, VGPRs 4-7 = K 16+8h..
            const int aoff = m * SP + cb2 + 4 * h;
            // B (32x16 f16, bias broadcast over N): lane(n,h) VGPRs 0-3 = K 8h..8h+7, VGPRs 4-7 = K 16+8h..
            const int boff = cb2 + 4 * h;
            union { uint4 q[2]; v16h v; } A, B;
            A.q[0] = *(const uint4*)&sgd[aoff];
            A.q[1] = *(const uint4*)&sgd[aoff + 8];
            B.q[0] = *(const uint4*)&sbd[boff];
            B.q[1] = *(const uint4*)&sbd[boff + 8];
            acc = __builtin_amdgcn_wmma_f32_16x16x32_f16(
                false, A.v, false, B.v, (short)0, acc, false, false);
        }
        __syncthreads();   // before next tile overwrites LDS
    }

    // D[m][n] identical across n; column 0 lives in lane0 (rows 0-7, vgpr r)
    // and lane16 (rows 8-15) per the 32-bit C/D layout.
    if (lane == 0) {
        #pragma unroll
        for (int r = 0; r < 8; ++r) spart[wave][r] = acc[r];
    } else if (lane == 16) {
        #pragma unroll
        for (int r = 0; r < 8; ++r) spart[wave][8 + r] = acc[r];
    }
    __syncthreads();
    if (t < ROWS_WG) {
        // deterministic split-K partials -> workspace (no float atomics)
        part[blockIdx.y * BATCH + rowbase + t] =
            spart[0][t] + spart[1][t] + spart[2][t] + spart[3][t];
    }
}

__global__ __launch_bounds__(1024) void lr_finalize(
    const float* __restrict__ part, const float* __restrict__ gbias,
    const float* __restrict__ label, float* __restrict__ out)
{
    __shared__ float red[1024];
    const int t = threadIdx.x;
    float lsum = 0.f;
    for (int b = t; b < BATCH; b += 1024) {
        float s = gbias[0];
        #pragma unroll
        for (int k = 0; k < SPLITK; ++k) s += part[k * BATCH + b];   // fixed order
        out[b]         = s;                       // logits
        out[BATCH + b] = 1.f / (1.f + expf(-s));  // pred
        const float y = label[b];
        lsum += fmaxf(s, 0.f) - s * y + log1pf(expf(-fabsf(s)));
    }
    red[t] = lsum;
    __syncthreads();
    for (int off = 512; off > 0; off >>= 1) {
        if (t < off) red[t] += red[t + off];
        __syncthreads();
    }
    if (t == 0) out[2 * BATCH] = red[0];          // loss
}

extern "C" void kernel_launch(void* const* d_in, const int* in_sizes, int n_in,
                              void* d_out, int out_size, void* d_ws, size_t ws_size,
                              hipStream_t stream) {
    const float* gate  = (const float*)d_in[0];  // [2048*65536]
    const float* bias  = (const float*)d_in[1];  // [65536]
    const float* gbias = (const float*)d_in[2];  // [1]
    const float* label = (const float*)d_in[3];  // [2048]
    float* out  = (float*)d_out;                 // [2048 logits | 2048 pred | 1 loss]
    float* part = (float*)d_ws;                  // SPLITK * BATCH f32 = 128KB

    dim3 grid(BATCH / ROWS_WG, SPLITK);          // 128 x 16 = 2048 workgroups
    lr_gemv_wmma<<<grid, 128, 0, stream>>>(gate, bias, part);
    lr_finalize<<<1, 1024, 0, stream>>>(part, gbias, label, out);
}
